// DeepMamba_60206851555644
// MI455X (gfx1250) — compile-verified, matrix operand order checked by reading, not run on hardware
//
#include <hip/hip_runtime.h>
#include <math.h>

// ---------------- model constants ----------------
#define NL    5
#define DM    1024
#define DI    2048          // EXPAND * D_MODEL
#define NST   16            // D_STATE
#define DTR   64            // DT_RANK
#define RDIM  (DTR + 2*NST) // 96
#define BATCH 4
#define LSEQ  2048
#define BL    (BATCH*LSEQ)  // 8192

typedef __attribute__((ext_vector_type(16))) __bf16 v16bf;
typedef __attribute__((ext_vector_type(8)))  float  v8f;
typedef __attribute__((ext_vector_type(4)))  int    v4i;

#if __has_builtin(__builtin_amdgcn_global_load_async_to_lds_b128) && \
    __has_builtin(__builtin_amdgcn_s_wait_asynccnt)
#define ASYNC_STAGE 1
#else
#define ASYNC_STAGE 0
#endif

#define AS1 __attribute__((address_space(1)))
#define AS3 __attribute__((address_space(3)))

__device__ __forceinline__ unsigned short f32_to_bf16(float x) {
  union { float f; unsigned u; } a; a.f = x;
  unsigned u = a.u;
  unsigned r = (u + 0x7FFFu + ((u >> 16) & 1u)) >> 16;   // round-to-nearest-even
  return (unsigned short)r;
}

// ---------------- generic f32 -> bf16 cast ----------------
__global__ __launch_bounds__(256) void k_cast_bf16(const float* __restrict__ src,
                                                   unsigned short* __restrict__ dst,
                                                   int n) {
  int i = blockIdx.x * 256 + threadIdx.x;
  if (i < n) dst[i] = f32_to_bf16(src[i]);
}

// ---------------- WMMA bf16 GEMM:  C[M,N] = A[M,K] * B[N,K]^T (+ addend) ----------------
// Block: 256 threads = 8 waves (4x2); block tile 128x128; wave tile 32x64 (2x4 frags).
// K step 32, LDS pitch 20 dwords (bank-conflict-free + 16B aligned -> ds_load_b128),
// double-buffered with gfx1250 async global->LDS copies when available.
#define LPITCH 20

__device__ __forceinline__ v16bf lds_frag(const unsigned* s, int rowBase, int lane) {
  int row  = rowBase + (lane & 15);
  int half = lane >> 4;
  const unsigned* p = s + row * LPITCH + half * 4;
  union { v16bf v; uint4 q[2]; } f;
  f.q[0] = *reinterpret_cast<const uint4*>(p);      // K = 8*half + {0..7}
  f.q[1] = *reinterpret_cast<const uint4*>(p + 8);  // K = 16 + 8*half + {0..7}
  return f.v;
}

__global__ __launch_bounds__(256) void k_gemm_bf16(
    const unsigned short* __restrict__ A, const unsigned short* __restrict__ B,
    float* __restrict__ C, const float* __restrict__ addend, int has_add,
    int M, int N, int K, int lda, int ldb, int ldc) {
  __shared__ unsigned sA[2][128 * LPITCH];
  __shared__ unsigned sB[2][128 * LPITCH];

  const int tid  = threadIdx.x;
  const int lane = tid & 31;
  const int wave = tid >> 5;
  const int wm   = wave >> 1;      // 0..3 -> row quadrant
  const int wn   = wave & 1;       // 0..1 -> col half
  const int m0   = blockIdx.y * 128;
  const int n0   = blockIdx.x * 128;

  const unsigned* Ad = reinterpret_cast<const unsigned*>(A);
  const unsigned* Bd = reinterpret_cast<const unsigned*>(B);
  const int ldaD = lda >> 1, ldbD = ldb >> 1;

  v8f acc[2][4];
  const v8f vzero = {0.f,0.f,0.f,0.f,0.f,0.f,0.f,0.f};
#pragma unroll
  for (int i = 0; i < 2; ++i)
#pragma unroll
    for (int j = 0; j < 4; ++j) acc[i][j] = vzero;

  const int r = tid >> 1;   // 0..127 tile row
  const int h = tid & 1;    // which 8-dword half of the 16-dword (K=32) row

  // clamp out-of-range rows to a valid row: the resulting garbage only feeds
  // D elements whose stores are guarded off (col >= N / row >= M).
  int ga_row = m0 + r; if (ga_row >= M) ga_row = M - 1;
  int gb_row = n0 + r; if (gb_row >= N) gb_row = N - 1;
  const unsigned* paBase = Ad + (size_t)ga_row * ldaD + h * 8;
  const unsigned* pbBase = Bd + (size_t)gb_row * ldbD + h * 8;
  const int ldst = r * LPITCH + h * 8;

  auto stage = [&](unsigned* dA, unsigned* dB, int k0) {
    const unsigned* pa = paBase + (k0 >> 1);
    const unsigned* pb = pbBase + (k0 >> 1);
    unsigned* la = dA + ldst;
    unsigned* lb = dB + ldst;
#if ASYNC_STAGE
    __builtin_amdgcn_global_load_async_to_lds_b128((AS1 v4i*)pa,       (AS3 v4i*)la,       0, 0);
    __builtin_amdgcn_global_load_async_to_lds_b128((AS1 v4i*)(pa + 4), (AS3 v4i*)(la + 4), 0, 0);
    __builtin_amdgcn_global_load_async_to_lds_b128((AS1 v4i*)pb,       (AS3 v4i*)lb,       0, 0);
    __builtin_amdgcn_global_load_async_to_lds_b128((AS1 v4i*)(pb + 4), (AS3 v4i*)(lb + 4), 0, 0);
#else
    uint4 qa0 = *reinterpret_cast<const uint4*>(pa);
    uint4 qa1 = *reinterpret_cast<const uint4*>(pa + 4);
    uint4 qb0 = *reinterpret_cast<const uint4*>(pb);
    uint4 qb1 = *reinterpret_cast<const uint4*>(pb + 4);
    *reinterpret_cast<uint4*>(la)     = qa0;
    *reinterpret_cast<uint4*>(la + 4) = qa1;
    *reinterpret_cast<uint4*>(lb)     = qb0;
    *reinterpret_cast<uint4*>(lb + 4) = qb1;
#endif
  };

  stage(sA[0], sB[0], 0);
  int ibuf = 0;
  for (int k0 = 0; k0 < K; k0 += 32) {
    const int nxt = k0 + 32;
    if (nxt < K) stage(sA[ibuf ^ 1], sB[ibuf ^ 1], nxt);
#if ASYNC_STAGE
    if (nxt < K) __builtin_amdgcn_s_wait_asynccnt(4);  // current tile retired
    else         __builtin_amdgcn_s_wait_asynccnt(0);
#endif
    __syncthreads();

    v16bf af[2], bfr[4];
    af[0] = lds_frag(sA[ibuf], wm * 32,      lane);
    af[1] = lds_frag(sA[ibuf], wm * 32 + 16, lane);
#pragma unroll
    for (int j = 0; j < 4; ++j) bfr[j] = lds_frag(sB[ibuf], wn * 64 + j * 16, lane);

#pragma unroll
    for (int fm = 0; fm < 2; ++fm)
#pragma unroll
      for (int fn = 0; fn < 4; ++fn)
        acc[fm][fn] = __builtin_amdgcn_wmma_f32_16x16x32_bf16(
            false, af[fm], false, bfr[fn], (short)0, acc[fm][fn], false, false);

    __syncthreads();
    ibuf ^= 1;
  }

  // --- epilogue: C/D layout — VGPR rr holds M=rr (lanes 0-15) / M=rr+8 (lanes 16-31) ---
#pragma unroll
  for (int fm = 0; fm < 2; ++fm) {
#pragma unroll
    for (int fn = 0; fn < 4; ++fn) {
      int col  = n0 + wn * 64 + fn * 16 + (lane & 15);
      int rowb = m0 + wm * 32 + fm * 16 + ((lane >> 4) << 3);
      if (col < N) {
#pragma unroll
        for (int rr = 0; rr < 8; ++rr) {
          int row = rowb + rr;
          if (row < M) {
            float v = acc[fm][fn][rr];
            if (has_add) v += addend[(size_t)row * ldc + col];
            C[(size_t)row * ldc + col] = v;
          }
        }
      }
    }
  }
}

// ---------------- depthwise causal conv (K=4) + SiLU ----------------
__global__ __launch_bounds__(256) void k_conv_silu(
    const float* __restrict__ xz, const float* __restrict__ cw,
    const float* __restrict__ cb, float* __restrict__ xa,
    unsigned short* __restrict__ xabf) {
  int idx = blockIdx.x * 256 + threadIdx.x;
  if (idx >= BL * DI) return;
  int d = idx % DI;
  int l = (idx / DI) % LSEQ;
  int b = idx / (DI * LSEQ);
  float acc = cb[d];
#pragma unroll
  for (int k = 0; k < 4; ++k) {
    int ls = l - 3 + k;
    if (ls >= 0)
      acc += cw[d * 4 + k] * xz[((size_t)(b * LSEQ + ls)) * (2 * DI) + d];
  }
  float s = acc / (1.f + __expf(-acc));   // SiLU
  xa[idx]   = s;
  xabf[idx] = f32_to_bf16(s);
}

// ---------------- softplus(delta_raw + b_dt) in place ----------------
__global__ __launch_bounds__(256) void k_softplus(float* __restrict__ delta,
                                                  const float* __restrict__ bdt) {
  int i = blockIdx.x * 256 + threadIdx.x;
  if (i >= BL * DI) return;
  float v = delta[i] + bdt[i % DI];
  delta[i] = (v > 20.f) ? v : log1pf(__expf(v));
}

// ---------------- selective scan ----------------
// thread = (b, d, n); 16 state lanes per d; xor-shuffle reduce over n; lane n==0
// applies D skip + SiLU(z) gate and stores y as bf16.
__global__ __launch_bounds__(256) void k_scan(
    const float* __restrict__ delta, const float* __restrict__ xa,
    const float* __restrict__ xdbl,  const float* __restrict__ xz,
    const float* __restrict__ Alog,  const float* __restrict__ Dp,
    unsigned short* __restrict__ ybf) {
  int t = blockIdx.x * 256 + threadIdx.x;     // BATCH*DI*NST threads
  int n  = t & 15;
  int bd = t >> 4;
  int d  = bd % DI;
  int b  = bd / DI;
  if (b >= BATCH) return;

  const float Aval = -__expf(Alog[d * NST + n]);
  const float Dv   = Dp[d];
  float h = 0.f;

  for (int l = 0; l < LSEQ; ++l) {
    size_t rowD = (size_t)(b * LSEQ + l);
    float dv = delta[rowD * DI + d];
    float u  = xa[rowD * DI + d];
    float Bt = xdbl[rowD * RDIM + DTR + n];
    float Ct = xdbl[rowD * RDIM + DTR + NST + n];
    h = __expf(dv * Aval) * h + dv * Bt * u;
    float part = h * Ct;
    part += __shfl_xor(part, 1, 32);
    part += __shfl_xor(part, 2, 32);
    part += __shfl_xor(part, 4, 32);
    part += __shfl_xor(part, 8, 32);
    if (n == 0) {
      float y = part + u * Dv;
      float z = xz[rowD * (2 * DI) + DI + d];
      y *= z / (1.f + __expf(-z));          // * SiLU(z)
      ybf[rowD * DI + d] = f32_to_bf16(y);
    }
  }
}

// ---------------- launcher ----------------
extern "C" void kernel_launch(void* const* d_in, const int* in_sizes, int n_in,
                              void* d_out, int out_size, void* d_ws, size_t ws_size,
                              hipStream_t stream) {
  const float* x    = (const float*)d_in[0];
  const float* Win  = (const float*)d_in[1];
  const float* cw   = (const float*)d_in[2];
  const float* cb   = (const float*)d_in[3];
  const float* Wx   = (const float*)d_in[4];
  const float* Wdt  = (const float*)d_in[5];
  const float* bdt  = (const float*)d_in[6];
  const float* Alog = (const float*)d_in[7];
  const float* Dpar = (const float*)d_in[8];
  const float* Wout = (const float*)d_in[9];

  char*  ws  = (char*)d_ws;
  size_t off = 0;
  auto alloc = [&](size_t bytes) -> void* {
    void* p = ws + off;
    off += (bytes + 255) & ~(size_t)255;
    return p;
  };

  float* xcA   = (float*)alloc((size_t)BL * DM * 4);
  float* xcB   = (float*)alloc((size_t)BL * DM * 4);
  float* xz    = (float*)alloc((size_t)BL * 2 * DI * 4);
  float* xa    = (float*)alloc((size_t)BL * DI * 4);
  float* delta = (float*)alloc((size_t)BL * DI * 4);
  float* xdbl  = (float*)alloc((size_t)BL * RDIM * 4);
  unsigned short* xbf    = (unsigned short*)alloc((size_t)BL * DM * 2);
  unsigned short* xabf   = (unsigned short*)alloc((size_t)BL * DI * 2);
  unsigned short* xdblbf = (unsigned short*)alloc((size_t)BL * RDIM * 2);
  unsigned short* ybf    = (unsigned short*)alloc((size_t)BL * DI * 2);
  unsigned short* winbf  = (unsigned short*)alloc((size_t)2 * DI * DM * 2);
  unsigned short* wxbf   = (unsigned short*)alloc((size_t)RDIM * DI * 2);
  unsigned short* wdtbf  = (unsigned short*)alloc((size_t)DI * DTR * 2);
  unsigned short* woutbf = (unsigned short*)alloc((size_t)DM * DI * 2);

  auto cast = [&](const float* s, unsigned short* d, int n) {
    k_cast_bf16<<<(n + 255) / 256, 256, 0, stream>>>(s, d, n);
  };
  auto gemm = [&](const unsigned short* A, const unsigned short* B, float* C,
                  const float* add, int M, int N, int K, int lda, int ldb, int ldc) {
    dim3 g((N + 127) / 128, (M + 127) / 128);
    k_gemm_bf16<<<g, 256, 0, stream>>>(A, B, C, add ? add : C, add ? 1 : 0,
                                       M, N, K, lda, ldb, ldc);
  };

  const float* xcur = x;
  for (int i = 0; i < NL; ++i) {
    const float* Wi  = Win  + (size_t)i * 2 * DI * DM;
    const float* cwi = cw   + (size_t)i * DI * 4;
    const float* cbi = cb   + (size_t)i * DI;
    const float* Wxi = Wx   + (size_t)i * RDIM * DI;
    const float* Wdi = Wdt  + (size_t)i * DI * DTR;
    const float* bdi = bdt  + (size_t)i * DI;
    const float* Ali = Alog + (size_t)i * DI * NST;
    const float* Dpi = Dpar + (size_t)i * DI;
    const float* Woi = Wout + (size_t)i * DM * DI;

    cast(Wi,  winbf,  2 * DI * DM);
    cast(Wxi, wxbf,   RDIM * DI);
    cast(Wdi, wdtbf,  DI * DTR);
    cast(Woi, woutbf, DM * DI);
    cast(xcur, xbf,   BL * DM);

    // xz = xcur @ W_in^T                 [8192 x 4096 x 1024]
    gemm(xbf, winbf, xz, nullptr, BL, 2 * DI, DM, DM, DM, 2 * DI);

    // depthwise conv + SiLU -> xa (f32 + bf16)
    k_conv_silu<<<(BL * DI + 255) / 256, 256, 0, stream>>>(xz, cwi, cbi, xa, xabf);

    // xdbl = xa @ W_x^T                  [8192 x 96 x 2048]
    gemm(xabf, wxbf, xdbl, nullptr, BL, RDIM, DI, DI, DI, RDIM);
    cast(xdbl, xdblbf, BL * RDIM);

    // delta_raw = dt @ W_dt^T            [8192 x 2048 x 64]
    gemm(xdblbf, wdtbf, delta, nullptr, BL, DI, DTR, RDIM, DTR, DI);
    k_softplus<<<(BL * DI + 255) / 256, 256, 0, stream>>>(delta, bdi);

    // selective scan -> y (bf16, gated)
    k_scan<<<(BATCH * DI * NST + 255) / 256, 256, 0, stream>>>(
        delta, xa, xdbl, xz, Ali, Dpi, ybf);

    // out = xcur + y @ W_out^T           [8192 x 1024 x 2048], residual fused
    float* dst = (i == NL - 1) ? (float*)d_out : ((i & 1) ? xcB : xcA);
    gemm(ybf, woutbf, dst, xcur, BL, DM, DI, DI, DI, DM);
    xcur = dst;
  }
}
